// CluttrVAE_38706245271662
// MI455X (gfx1250) — compile-verified, matrix operand order checked by reading, not run on hardware
//
#include <hip/hip_runtime.h>
#include <hip/hip_bf16.h>

// ---------------------------------------------------------------------------
// Types for CDNA5 WMMA
// ---------------------------------------------------------------------------
typedef __attribute__((ext_vector_type(16))) _Float16 v16h;
typedef __attribute__((ext_vector_type(8)))  float    v8f;

#define BM 64
#define BN 128
#define TK 32
#define LSTRIDE 36   // LDS row stride in halves: 8-byte aligned rows (36*2=72)

struct GemmP {
    const float* A1; long lda1; const float* W1; int K1;   // pair 0
    const float* A2; long lda2; const float* W2; int K2;   // pair 1 (nullable)
    const float* Cin; int ldcin;                           // optional full-matrix add
    const float* bias;                                     // optional bias[N]
    float* C; int ldc;
    int M, N;
    int act;   // 0 none, 1 relu, 2 sigmoid, 3 tanh*6, 4 clip[-10,4]
};

// ---------------------------------------------------------------------------
// Generic WMMA GEMM:  C = act( A1@W1 [+ A2@W2] + bias + Cin )
// f32 in memory, f16 in the matrix pipes, f32 accumulation.
// Workgroup = 256 threads (8 waves), tile 64(M) x 128(N), K tiles of 32.
// Each wave computes a 32x32 block as 2x2 WMMA 16x16 fragments.
// Interior tiles use vectorized float4 staging; edge tiles use guarded path.
// ---------------------------------------------------------------------------
__global__ __launch_bounds__(256) void gemm_wmma(GemmP p0, GemmP p1) {
    const GemmP p = (blockIdx.z == 0) ? p0 : p1;

    __shared__ _Float16 As[BM][LSTRIDE];
    __shared__ _Float16 Wt[BN][LSTRIDE];

    const int tid  = threadIdx.x;
    const int lane = tid & 31;
    const int w    = tid >> 5;
    const int wm   = w >> 2;          // 2 waves along M
    const int wn   = w & 3;           // 4 waves along N
    const int lr   = lane & 15;
    const int lh   = (lane >> 4) & 1;
    const long mBase = (long)blockIdx.y * BM;
    const int  nBase = blockIdx.x * BN;
    const bool fullN = (nBase + BN <= p.N) && ((p.N & 3) == 0);

    v8f acc[2][2];
    for (int a = 0; a < 2; ++a)
        for (int b = 0; b < 2; ++b)
            acc[a][b] = v8f{};

    for (int pair = 0; pair < 2; ++pair) {
        const float* A  = pair ? p.A2  : p.A1;
        const float* W  = pair ? p.W2  : p.W1;
        const long  lda = pair ? p.lda2 : p.lda1;
        const int   K   = pair ? p.K2  : p.K1;
        if (A == nullptr) continue;
        const bool ldaOk = ((lda & 3) == 0);

        for (int k0 = 0; k0 < K; k0 += TK) {
            const bool fullK = (k0 + TK <= K);
            __syncthreads();

            if (fullK && ldaOk) {
                // ---- fast path A: 2x float4 per thread, ds_store_b64
                #pragma unroll
                for (int it = 0; it < 2; ++it) {
                    int vi = tid + it * 256;         // 0..511 float4 slots
                    int r  = vi >> 3;                // 64 rows, 8 float4/row
                    int c4 = (vi & 7) << 2;
                    const float4 v = *(const float4*)(A + (mBase + r) * lda + k0 + c4);
                    union { _Float16 h[4]; unsigned long long u; } pk;
                    pk.h[0] = (_Float16)v.x; pk.h[1] = (_Float16)v.y;
                    pk.h[2] = (_Float16)v.z; pk.h[3] = (_Float16)v.w;
                    *(unsigned long long*)&As[r][c4] = pk.u;
                }
            } else {
                // ---- guarded path A (K edge)
                #pragma unroll
                for (int i = 0; i < (BM * TK) / 256; ++i) {
                    int li = tid + i * 256;
                    int r  = li >> 5;
                    int c  = li & 31;
                    float v = 0.f;
                    int gk = k0 + c;
                    if (gk < K) v = A[(mBase + r) * lda + gk];
                    As[r][c] = (_Float16)v;
                }
            }

            if (fullK && fullN) {
                // ---- fast path W: 2(k) x 4(n) subtile per thread,
                //      transposed k-pair u32 stores into LDS
                #pragma unroll
                for (int it = 0; it < 2; ++it) {
                    int li = tid + it * 256;         // 0..511 subtiles
                    int kk = (li >> 5) << 1;         // even k in [0,32)
                    int nb = (li & 31) << 2;         // n base in [0,128)
                    const float4 v0 = *(const float4*)(W + (long)(k0 + kk) * p.N + nBase + nb);
                    const float4 v1 = *(const float4*)(W + (long)(k0 + kk + 1) * p.N + nBase + nb);
                    union { _Float16 h[2]; unsigned int u; } q;
                    q.h[0] = (_Float16)v0.x; q.h[1] = (_Float16)v1.x;
                    *(unsigned int*)&Wt[nb + 0][kk] = q.u;
                    q.h[0] = (_Float16)v0.y; q.h[1] = (_Float16)v1.y;
                    *(unsigned int*)&Wt[nb + 1][kk] = q.u;
                    q.h[0] = (_Float16)v0.z; q.h[1] = (_Float16)v1.z;
                    *(unsigned int*)&Wt[nb + 2][kk] = q.u;
                    q.h[0] = (_Float16)v0.w; q.h[1] = (_Float16)v1.w;
                    *(unsigned int*)&Wt[nb + 3][kk] = q.u;
                }
            } else {
                // ---- guarded path W (K or N edge)
                #pragma unroll
                for (int i = 0; i < (TK * BN) / 256; ++i) {
                    int li = tid + i * 256;
                    int kk = li >> 7;
                    int n  = li & 127;
                    float v = 0.f;
                    int gk = k0 + kk, gn = nBase + n;
                    if (gk < K && gn < p.N) v = W[(long)gk * p.N + gn];
                    Wt[n][kk] = (_Float16)v;
                }
            }

            // prefetch next K tile of A into cache (global_prefetch_b8)
            if (k0 + TK < K)
                __builtin_prefetch(A + (mBase + (tid & 63)) * lda + k0 + TK, 0, 1);
            __syncthreads();

            // ---- build fragments per the 16-bit A 16x32 VGPR layout
            union F { v16h v; _Float16 h[16]; };
            F af[2], bf[2];
            #pragma unroll
            for (int mi = 0; mi < 2; ++mi) {
                int r = wm * 32 + mi * 16 + lr;
                #pragma unroll
                for (int vv = 0; vv < 8; ++vv) {
                    int k = ((vv >> 2) << 4) + (lh << 3) + ((vv & 3) << 1);
                    af[mi].h[2 * vv]     = As[r][k];
                    af[mi].h[2 * vv + 1] = As[r][k + 1];
                }
            }
            #pragma unroll
            for (int ni = 0; ni < 2; ++ni) {
                int c = wn * 32 + ni * 16 + lr;
                #pragma unroll
                for (int vv = 0; vv < 8; ++vv) {
                    int k = ((vv >> 2) << 4) + (lh << 3) + ((vv & 3) << 1);
                    bf[ni].h[2 * vv]     = Wt[c][k];
                    bf[ni].h[2 * vv + 1] = Wt[c][k + 1];
                }
            }
            #pragma unroll
            for (int mi = 0; mi < 2; ++mi)
                #pragma unroll
                for (int ni = 0; ni < 2; ++ni)
                    acc[mi][ni] = __builtin_amdgcn_wmma_f32_16x16x32_f16(
                        false, af[mi].v, false, bf[ni].v,
                        (short)0, acc[mi][ni], false, false);
        }
    }

    // ---- writeback with fused bias / Cin / activation
    #pragma unroll
    for (int mi = 0; mi < 2; ++mi) {
        #pragma unroll
        for (int ni = 0; ni < 2; ++ni) {
            int n = nBase + wn * 32 + ni * 16 + lr;
            if (n >= p.N) continue;
            long mrow = mBase + wm * 32 + mi * 16 + lh * 8;
            float bv = p.bias ? p.bias[n] : 0.f;
            #pragma unroll
            for (int j = 0; j < 8; ++j) {
                long m = mrow + j;
                float v = acc[mi][ni][j] + bv;
                if (p.Cin) v += p.Cin[m * p.ldcin + n];
                switch (p.act) {
                    case 1: v = v > 0.f ? v : 0.f; break;
                    case 2: v = 1.f / (1.f + expf(-v)); break;
                    case 3: v = tanhf(v) * 6.f; break;
                    case 4: v = v < -10.f ? -10.f : (v > 4.f ? 4.f : v); break;
                    default: break;
                }
                p.C[m * p.ldc + n] = v;
            }
        }
    }
}

// ---------------------------------------------------------------------------
// Elementwise / helper kernels
// ---------------------------------------------------------------------------
__global__ void embed_kernel(const int* __restrict__ x,
                             const float* __restrict__ emb,
                             float* __restrict__ out, long total) {
    long i = (long)blockIdx.x * 256 + threadIdx.x;
    if (i >= total) return;
    long r = i / 300;
    int  c = (int)(i % 300);
    out[i] = emb[(long)x[r] * 300 + c];
}

__global__ void hw_combine(const float* __restrict__ gate,
                           const float* __restrict__ fg,
                           const float* __restrict__ q2,
                           float* __restrict__ out, long total) {
    long i = (long)blockIdx.x * 256 + threadIdx.x;
    if (i >= total) return;
    float g = gate[i];
    out[i] = g * fg[i] + (1.f - g) * q2[i];
}

__global__ void concat_henc(const float* __restrict__ hf,
                            const float* __restrict__ hb,
                            float* __restrict__ henc) {
    int i = blockIdx.x * 256 + threadIdx.x;
    if (i >= 64 * 600) return;
    int m = i / 600, c = i % 600;
    henc[i] = (c < 300) ? hf[m * 300 + c] : hb[m * 300 + (c - 300)];
}

__global__ void z_compute(const float* __restrict__ mean,
                          const float* __restrict__ logvar,
                          const float* __restrict__ eps,
                          float* __restrict__ z) {
    int i = blockIdx.x * 256 + threadIdx.x;
    if (i >= 64 * 64) return;
    z[i] = mean[i] + eps[i] * expf(0.5f * logvar[i]);
}

__global__ void lstm_point(const float* __restrict__ gF,
                           const float* __restrict__ gB,
                           float* hF, float* cF, float* hB, float* cB,
                           int H, float* seqout, int t, int outW) {
    int dir = blockIdx.y;
    const float* g = dir ? gB : gF;
    float* h = dir ? hB : hF;
    float* c = dir ? cB : cF;
    int i = blockIdx.x * 256 + threadIdx.x;
    if (i >= 64 * H) return;
    int m = i / H, n = i % H;
    const float* row = g + (long)m * 4 * H;
    float ig = row[n], fg = row[H + n], gg = row[2 * H + n], og = row[3 * H + n];
    float cs = (1.f / (1.f + expf(-fg))) * c[i] +
               (1.f / (1.f + expf(-ig))) * tanhf(gg);
    float hs = (1.f / (1.f + expf(-og))) * tanhf(cs);
    c[i] = cs;
    h[i] = hs;
    if (seqout) {
        int te = dir ? (512 - 1 - t) : t;
        seqout[(long)m * 512 * outW + (long)te * outW + dir * H + n] = hs;
    }
}

__global__ void zero_kernel(float* p, long n) {
    long i = (long)blockIdx.x * 256 + threadIdx.x;
    if (i < n) p[i] = 0.f;
}

// ---------------------------------------------------------------------------
// Host orchestration
// ---------------------------------------------------------------------------
extern "C" void kernel_launch(void* const* d_in, const int* in_sizes, int n_in,
                              void* d_out, int out_size, void* d_ws, size_t ws_size,
                              hipStream_t stream) {
    (void)in_sizes; (void)n_in; (void)out_size; (void)ws_size;

    const int SEQ = 512, Bz = 64;
    const long ROWS = (long)Bz * SEQ;          // 32768

    // ---- inputs (setup_inputs dict order, recursively flattened)
    const int*   x   = (const int*)d_in[0];
    const float* eps = (const float*)d_in[1];
    const float* emb = (const float*)d_in[2];
    const float *hwW[2][5], *hwB[2][5];
    int base = 3;
    for (int h = 0; h < 2; ++h)
        for (int k = 0; k < 5; ++k) {
            hwW[h][k] = (const float*)d_in[base++];
            hwB[h][k] = (const float*)d_in[base++];
        }
    const float* encfWx = (const float*)d_in[23];
    const float* encfWh = (const float*)d_in[24];
    const float* encfB  = (const float*)d_in[25];
    const float* encbWx = (const float*)d_in[26];
    const float* encbWh = (const float*)d_in[27];
    const float* encbB  = (const float*)d_in[28];
    const float* meanW  = (const float*)d_in[29];
    const float* meanB  = (const float*)d_in[30];
    const float* lvW    = (const float*)d_in[31];
    const float* lvB    = (const float*)d_in[32];
    const float* d1fWx  = (const float*)d_in[33];
    const float* d1fWh  = (const float*)d_in[34];
    const float* d1fB   = (const float*)d_in[35];
    const float* d1bWx  = (const float*)d_in[36];
    const float* d1bWh  = (const float*)d_in[37];
    const float* d1bB   = (const float*)d_in[38];
    const float* d2fWx  = (const float*)d_in[39];
    const float* d2fWh  = (const float*)d_in[40];
    const float* d2fB   = (const float*)d_in[41];
    const float* d2bWx  = (const float*)d_in[42];
    const float* d2bWh  = (const float*)d_in[43];
    const float* d2bB   = (const float*)d_in[44];
    const float* outW   = (const float*)d_in[45];
    const float* outB   = (const float*)d_in[46];

    // ---- outputs
    float* logits  = (float*)d_out;
    float* meanO   = logits + (long)ROWS * 170;
    float* logvarO = meanO + 64 * 64;

    // ---- workspace layout (floats)
    float* ws = (float*)d_ws;
    long cur = 0;
    auto alloc = [&](long n) { float* p = ws + cur; cur += n; return p; };
    float* ACT = alloc(ROWS * 300);
    float* T1  = alloc(ROWS * 300);
    float* T2  = alloc(ROWS * 300);
    float* T3  = alloc(ROWS * 300);
    float* D1  = alloc(ROWS * 800);
    float* D2  = alloc(ROWS * 800);
    float* HENC = alloc(64 * 600);
    float* Z    = alloc(64 * 64);
    float* ZXWF = alloc(64 * 1600);
    float* ZXWB = alloc(64 * 1600);
    float* STATES = alloc(0);
    float* HEF = alloc(64 * 300); float* CEF = alloc(64 * 300);
    float* HEB = alloc(64 * 300); float* CEB = alloc(64 * 300);
    float* H1F = alloc(64 * 400); float* C1F = alloc(64 * 400);
    float* H1B = alloc(64 * 400); float* C1B = alloc(64 * 400);
    float* H2F = alloc(64 * 400); float* C2F = alloc(64 * 400);
    float* H2B = alloc(64 * 400); float* C2B = alloc(64 * 400);
    long stateN = (ws + cur) - STATES;
    float* GF = alloc(64 * 1600);
    float* GB = alloc(64 * 1600);

    auto gemm1 = [&](const float* A, long lda, const float* W, int K,
                     const float* bias, const float* Cin, int ldcin,
                     float* C, int ldc, int M, int N, int act) {
        GemmP p{A, lda, W, K, nullptr, 0, nullptr, 0, Cin, ldcin, bias, C, ldc, M, N, act};
        dim3 g((N + BN - 1) / BN, M / BM, 1);
        gemm_wmma<<<g, 256, 0, stream>>>(p, p);
    };
    auto gemm_pair = [&](const GemmP& a, const GemmP& b) {
        dim3 g((a.N + BN - 1) / BN, a.M / BM, 2);
        gemm_wmma<<<g, 256, 0, stream>>>(a, b);
    };

    // ---- 0. zero all LSTM states
    zero_kernel<<<(unsigned)((stateN + 255) / 256), 256, 0, stream>>>(STATES, stateN);

    // ---- 1. embedding gather -> ACT [32768, 300]
    {
        long tot = ROWS * 300;
        embed_kernel<<<(unsigned)((tot + 255) / 256), 256, 0, stream>>>(x, emb, ACT, tot);
    }

    // ---- 2. two highway layers (ACT -> ACT)
    for (int h = 0; h < 2; ++h) {
        long tot = ROWS * 300;
        gemm1(ACT, 300, hwW[h][0], 300, hwB[h][0], nullptr, 0, T1, 300, ROWS, 300, 1); // g=relu
        gemm1(T1, 300, hwW[h][1], 300, hwB[h][1], nullptr, 0, T2, 300, ROWS, 300, 1);  // fg=relu
        gemm1(ACT, 300, hwW[h][2], 300, hwB[h][2], nullptr, 0, T1, 300, ROWS, 300, 1); // q1=relu
        gemm1(T1, 300, hwW[h][3], 300, hwB[h][3], nullptr, 0, T3, 300, ROWS, 300, 0);  // q2
        gemm1(ACT, 300, hwW[h][4], 300, hwB[h][4], nullptr, 0, T1, 300, ROWS, 300, 2); // gate=sigmoid
        hw_combine<<<(unsigned)((tot + 255) / 256), 256, 0, stream>>>(T1, T2, T3, ACT, tot);
    }

    // ---- 3. encoder biLSTM (only final hidden states needed)
    for (int t = 0; t < SEQ; ++t) {
        GemmP pf{ACT + (long)t * 300, (long)SEQ * 300, encfWx, 300,
                 HEF, 300, encfWh, 300, nullptr, 0, encfB, GF, 1200, 64, 1200, 0};
        GemmP pb{ACT + (long)(SEQ - 1 - t) * 300, (long)SEQ * 300, encbWx, 300,
                 HEB, 300, encbWh, 300, nullptr, 0, encbB, GB, 1200, 64, 1200, 0};
        gemm_pair(pf, pb);
        dim3 g((64 * 300 + 255) / 256, 2);
        lstm_point<<<g, 256, 0, stream>>>(GF, GB, HEF, CEF, HEB, CEB, 300,
                                          nullptr, t, 0);
    }
    concat_henc<<<(64 * 600 + 255) / 256, 256, 0, stream>>>(HEF, HEB, HENC);

    // ---- 4. latent: mean = tanh(.)*6, logvar = clip(., -10, 4), z = mean+eps*exp(lv/2)
    gemm1(HENC, 600, meanW, 600, meanB, nullptr, 0, meanO, 64, 64, 64, 3);
    gemm1(HENC, 600, lvW, 600, lvB, nullptr, 0, logvarO, 64, 64, 64, 4);
    z_compute<<<(64 * 64 + 255) / 256, 256, 0, stream>>>(meanO, logvarO, eps, Z);

    // ---- 5. decoder 1: input is z for every step -> precompute z@Wx+b once per dir
    {
        GemmP pf{Z, 64, d1fWx, 64, nullptr, 0, nullptr, 0, nullptr, 0, d1fB,
                 ZXWF, 1600, 64, 1600, 0};
        GemmP pb{Z, 64, d1bWx, 64, nullptr, 0, nullptr, 0, nullptr, 0, d1bB,
                 ZXWB, 1600, 64, 1600, 0};
        gemm_pair(pf, pb);
    }
    for (int t = 0; t < SEQ; ++t) {
        GemmP pf{H1F, 400, d1fWh, 400, nullptr, 0, nullptr, 0, ZXWF, 1600,
                 nullptr, GF, 1600, 64, 1600, 0};
        GemmP pb{H1B, 400, d1bWh, 400, nullptr, 0, nullptr, 0, ZXWB, 1600,
                 nullptr, GB, 1600, 64, 1600, 0};
        gemm_pair(pf, pb);
        dim3 g((64 * 400 + 255) / 256, 2);
        lstm_point<<<g, 256, 0, stream>>>(GF, GB, H1F, C1F, H1B, C1B, 400,
                                          D1, t, 800);
    }

    // ---- 6. decoder 2: input D1 [B,S,800]
    for (int t = 0; t < SEQ; ++t) {
        GemmP pf{D1 + (long)t * 800, (long)SEQ * 800, d2fWx, 800,
                 H2F, 400, d2fWh, 400, nullptr, 0, d2fB, GF, 1600, 64, 1600, 0};
        GemmP pb{D1 + (long)(SEQ - 1 - t) * 800, (long)SEQ * 800, d2bWx, 800,
                 H2B, 400, d2bWh, 400, nullptr, 0, d2bB, GB, 1600, 64, 1600, 0};
        gemm_pair(pf, pb);
        dim3 g((64 * 400 + 255) / 256, 2);
        lstm_point<<<g, 256, 0, stream>>>(GF, GB, H2F, C2F, H2B, C2B, 400,
                                          D2, t, 800);
    }

    // ---- 7. output projection -> logits [32768, 170]
    gemm1(D2, 800, outW, 800, outB, nullptr, 0, logits, 170, ROWS, 170, 0);
}
